// FlowModel_19868518711466
// MI455X (gfx1250) — matrix operand order repaired and checked by reading, not compiled, not run on hardware
//
#include <hip/hip_runtime.h>
#include <hip/hip_bf16.h>

typedef __attribute__((ext_vector_type(16))) _Float16 v16h;
typedef __attribute__((ext_vector_type(8)))  _Float16 v8h;
typedef __attribute__((ext_vector_type(8)))  float    v8f;
typedef _Float16 half_t;

#define Bc   4
#define Nc   10000
#define Dc   16
#define EMBc 32
#define Fc   2
#define ENCc 64
#define Hc   4
#define HDc  32
#define BIGc 1e9f
#define TPB_NODE 625   /* node tiles per batch: N/16 */

// --- WMMA operand index maps (CDNA5 16x16x32 f16 layouts, wave32) ---
// A (16x32, MxK): lane<16 -> row M=lane, elems 0..7 => K 0..7, 8..15 => K 16..23
//                 lane>=16 -> same rows, K ranges offset by 8 / 24.
// B (32x16, KxN): lane holds column N=lane&15; same K map as A: elems 0..7 => K kb..kb+7, 8..15 => K kb+16..kb+23 (kb=0 / 8)
__device__ __forceinline__ int b_kmap(int lane, int e) {
  int kb = (lane < 16) ? 0 : 8;
  return (e < 8) ? (kb + e) : (kb + 8 + e);
}

__device__ __forceinline__ v8f wmma_f16(v16h a, v16h b, v8f c) {
  // 8 args: (neg_a, A, neg_b, B, c_mod, C, reuse_a, reuse_b)
  return __builtin_amdgcn_wmma_f32_16x16x32_f16(false, a, false, b, (short)0, c, false, false);
}

// A fragment from a contiguous f16 row (stride>=32 halfs): two b128 loads + shuffle
__device__ __forceinline__ v16h load_a_f16row(const half_t* __restrict__ row, int lane) {
  int kb = (lane < 16) ? 0 : 8;
  v8h lo = *(const v8h*)(row + kb);        // K = kb .. kb+7
  v8h hi = *(const v8h*)(row + kb + 16);   // K = kb+16 .. kb+23
  return __builtin_shufflevector(lo, hi, 0,1,2,3,4,5,6,7,8,9,10,11,12,13,14,15);
}

// A fragment from a contiguous f32 LDS row: vector LDS loads + cvt
__device__ __forceinline__ v16h load_a_f32row(const float* __restrict__ row, int lane) {
  int kb = (lane < 16) ? 0 : 8;
  v16h a;
#pragma unroll
  for (int e = 0; e < 8; ++e) a[e] = (half_t)row[kb + e];
#pragma unroll
  for (int e = 0; e < 8; ++e) a[8 + e] = (half_t)row[kb + 16 + e];
  return a;
}

// Pre-swizzle a row-major weight [Krows x Ncols] into WMMA B fragments:
// frag[((ks*NT+nt)*32+lane)*16+e] = W[(ks*32+b_kmap(lane,e)) * Ncols + nt*16+(lane&15)]
__device__ __forceinline__ void build_frags(const float* __restrict__ W, int Krows, int Ncols,
                                            int KS, int NT, half_t* __restrict__ frag, int t) {
  int total = KS * NT * 512;
  for (int i = t; i < total; i += 256) {
    int e = i & 15, lane = (i >> 4) & 31;
    int f = i >> 9;
    int nt = f % NT, ks = f / NT;
    int K = ks * 32 + b_kmap(lane, e);
    int col = nt * 16 + (lane & 15);
    frag[i] = (K < Krows) ? (half_t)W[K * Ncols + col] : (half_t)0.f;
  }
}

// ---------------------------------------------------------------- prep
__global__ void prep_kernel(const float* __restrict__ Wg, const float* __restrict__ a_src,
                            const float* __restrict__ a_tgt, const float* __restrict__ W_enc,
                            const float* __restrict__ Wd1, const float* __restrict__ Wdd1,
                            float* svec, float* tvec, half_t* wenc_frag, half_t* wg_frag,
                            half_t* wd1_frag, half_t* wdd1_frag) {
  int t = threadIdx.x; // 256 threads
  { // src_vec/tgt_vec [H][ENC] = einsum('heo,ho->he')
    int h = t >> 6, e = t & 63;
    float ss = 0.f, tt = 0.f;
    for (int o = 0; o < ENCc; ++o) {
      float w = Wg[(h * ENCc + e) * ENCc + o];
      ss += w * a_src[h * ENCc + o];
      tt += w * a_tgt[h * ENCc + o];
    }
    svec[t] = ss; tvec[t] = tt;
  }
  build_frags(W_enc, EMBc + Fc, ENCc, 2, 4, wenc_frag, t);  // [34,64] K-padded to 64
  build_frags(Wg,    256,       ENCc, 8, 4, wg_frag,   t);  // [256,64]
  build_frags(Wd1,   128,       HDc,  4, 2, wd1_frag,  t);  // [128,32]
  build_frags(Wdd1,  ENCc,      HDc,  2, 2, wdd1_frag, t);  // [64,32]
}

// packed f16 encoder input row [B*N][64]: 32 normalized-emb + 2 feat + 30 zeros
__global__ void build_xin_kernel(const float* __restrict__ emb, const float* __restrict__ feat,
                                 half_t* __restrict__ xin) {
  int idx = blockIdx.x * blockDim.x + threadIdx.x;
  if (idx >= Bc * Nc) return;
  int n = idx % Nc;
  float ev[EMBc]; float s = 0.f;
#pragma unroll
  for (int i = 0; i < EMBc; ++i) { ev[i] = emb[n * EMBc + i]; s += ev[i] * ev[i]; }
  float sc = 1.f / fmaxf(sqrtf(s), 1.f);   // max_norm = 1
  half_t row[64];
#pragma unroll
  for (int i = 0; i < EMBc; ++i) row[i] = (half_t)(ev[i] * sc);
  row[32] = (half_t)feat[idx * Fc + 0];
  row[33] = (half_t)feat[idx * Fc + 1];
#pragma unroll
  for (int i = 34; i < 64; ++i) row[i] = (half_t)0.f;
  v8h* dst = (v8h*)(xin + (size_t)idx * 64);
  const v8h* srcv = (const v8h*)row;
#pragma unroll
  for (int c = 0; c < 8; ++c) dst[c] = srcv[c];
}

// f32 -> f16 mirror of the padded state (8 elems per thread, b128 in / b128 out)
__global__ void x_to_half_kernel(const float* __restrict__ x, half_t* __restrict__ xh) {
  int c = blockIdx.x * blockDim.x + threadIdx.x;
  if (c >= Bc * (Nc + 1) * 8) return;
  const float* src = x + (size_t)c * 8;
  v8h o;
#pragma unroll
  for (int i = 0; i < 8; ++i) o[i] = (half_t)src[i];
  *(v8h*)(xh + (size_t)c * 8) = o;
}

__global__ void zero_f_kernel(float* p, int n) {
  int i = blockIdx.x * blockDim.x + threadIdx.x;
  if (i < n) p[i] = 0.f;
}

// zero the "pad row" (index N) of padded buffers so gathers of adj==N read zeros
__global__ void zero_pads_kernel(float* x0, float* x1, float* s_tgt, float* dual_pad) {
  int t = blockIdx.x * blockDim.x + threadIdx.x;
  if (t < 256)      { int b = t >> 6, e = t & 63; x0[((size_t)b * (Nc + 1) + Nc) * ENCc + e] = 0.f; }
  else if (t < 512) { int i = t - 256; int b = i >> 6, e = i & 63; x1[((size_t)b * (Nc + 1) + Nc) * ENCc + e] = 0.f; }
  else if (t < 528) { int i = t - 512; int b = i >> 2, h = i & 3;  s_tgt[((size_t)b * (Nc + 1) + Nc) * Hc + h] = 0.f; }
  else if (t < 532) { int b = t - 528; dual_pad[(size_t)b * (Nc + 1) + Nc] = 0.f; }
}

// ---------------------------------------------------------------- encoder: [B*N,34->64]@[64,64]
// 16 | N, so each 16-row tile lives in one batch: b = tile/625, n0 = (tile%625)*16
__global__ void encoder_kernel(const half_t* __restrict__ xin, const half_t* __restrict__ wfrag,
                               const float* __restrict__ b_enc, float* __restrict__ x0) {
  int lane = threadIdx.x & 31, wid = threadIdx.x >> 5;
  int tile = blockIdx.x * 8 + wid;
  if (tile >= Bc * TPB_NODE) return;         // wave-uniform guard: EXEC stays full
  int r16 = lane & 15;
  int b = tile / TPB_NODE;
  int n0 = (tile - b * TPB_NODE) * 16;
  const half_t* row = xin + ((size_t)(b * Nc + n0 + r16)) * 64;
  const v16h* bf = (const v16h*)wfrag;
  v8f acc[4] = {};
#pragma unroll
  for (int ks = 0; ks < 2; ++ks) {
    v16h a = load_a_f16row(row + ks * 32, lane);
#pragma unroll
    for (int nt = 0; nt < 4; ++nt)
      acc[nt] = wmma_f16(a, bf[(ks * 4 + nt) * 32 + lane], acc[nt]);
  }
  size_t obase = ((size_t)b * (Nc + 1) + n0) * ENCc;
#pragma unroll
  for (int nt = 0; nt < 4; ++nt) {
    int col = nt * 16 + r16;
    float bias = b_enc[col];
#pragma unroll
    for (int r = 0; r < 8; ++r) {
      int rowt = r + ((lane < 16) ? 0 : 8);
      x0[obase + (size_t)rowt * ENCc + col] = acc[nt][r] + bias;
    }
  }
}

// ---------------------------------------------------------------- GAT
__global__ void gat_scores_kernel(const float* __restrict__ x, const float* __restrict__ svec,
                                  const float* __restrict__ tvec, float* s_src, float* s_tgt) {
  int idx = blockIdx.x * blockDim.x + threadIdx.x;
  if (idx >= Bc * Nc) return;
  int b = idx / Nc, n = idx % Nc;
  const float* xr = x + ((size_t)b * (Nc + 1) + n) * ENCc;
  for (int h = 0; h < Hc; ++h) {
    float ss = 0.f, tt = 0.f;
    for (int e = 0; e < ENCc; ++e) { float v = xr[e]; ss += v * svec[h * ENCc + e]; tt += v * tvec[h * ENCc + e]; }
    s_src[idx * Hc + h] = ss;
    s_tgt[((size_t)b * (Nc + 1) + n) * Hc + h] = tt;
  }
}

__global__ void gat_attn_kernel(const float* __restrict__ s_src, const float* __restrict__ s_tgt,
                                const int* __restrict__ adj, const int* __restrict__ numn,
                                float* __restrict__ alpha) {
  int idx = blockIdx.x * blockDim.x + threadIdx.x; // B*N*H, exact multiple of 256
  int h = idx & 3, bn = idx >> 2;
  int b = bn / Nc;
  float ss = s_src[bn * Hc + h];
  int nn = numn[b];
  float lo[Dc]; float mx = -3.0e38f;
  for (int d = 0; d < Dc; ++d) {
    int av = adj[bn * Dc + d];
    float l;
    if (av == nn) l = -BIGc;
    else { float z = ss + s_tgt[((size_t)b * (Nc + 1) + av) * Hc + h]; l = (z > 0.f) ? z : 0.2f * z; }
    lo[d] = l; mx = fmaxf(mx, l);
  }
  float sum = 0.f;
  for (int d = 0; d < Dc; ++d) { float e2 = __expf(lo[d] - mx); lo[d] = e2; sum += e2; }
  float inv = 1.f / sum;
  for (int d = 0; d < Dc; ++d) alpha[((size_t)bn * Dc + d) * Hc + h] = lo[d] * inv;
}

// per-wave: build agg[16,256] in LDS, then WMMA project @ Wg[256,64], tanh(./H)
__global__ void gat_aggregate_kernel(const float* __restrict__ x, const float* __restrict__ alpha,
                                     const int* __restrict__ adj, const half_t* __restrict__ wg_frag,
                                     float* __restrict__ xn) {
  __shared__ float agg[2][16][Hc * ENCc];   // 32 KB
  int lane = threadIdx.x & 31, wid = threadIdx.x >> 5;
  int tile = blockIdx.x * 2 + wid;          // 1250*2 = 2500 tiles, exact
  int b = tile / TPB_NODE;
  int n0 = (tile - b * TPB_NODE) * 16;
  for (int i = lane; i < 16 * Hc * ENCc; i += 32) agg[wid][i >> 8][i & 255] = 0.f;
  __syncthreads();
  int r16 = lane & 15;
  size_t xbase = (size_t)b * (Nc + 1);
  for (int j = 0; j < 16; ++j) {
    int bn = b * Nc + n0 + j;
    float* ag = agg[wid][j];
    for (int d = 0; d < Dc; ++d) {
      int av = adj[bn * Dc + d];
      const float* al = &alpha[((size_t)bn * Dc + d) * Hc];
      float a0 = al[0], a1 = al[1], a2 = al[2], a3 = al[3];
      const float* nb = &x[(xbase + av) * ENCc];
      float v0 = nb[2 * lane], v1 = nb[2 * lane + 1];
      ag[0 * ENCc + 2 * lane] += a0 * v0; ag[0 * ENCc + 2 * lane + 1] += a0 * v1;
      ag[1 * ENCc + 2 * lane] += a1 * v0; ag[1 * ENCc + 2 * lane + 1] += a1 * v1;
      ag[2 * ENCc + 2 * lane] += a2 * v0; ag[2 * ENCc + 2 * lane + 1] += a2 * v1;
      ag[3 * ENCc + 2 * lane] += a3 * v0; ag[3 * ENCc + 2 * lane + 1] += a3 * v1;
    }
  }
  __syncthreads();
  const v16h* bf = (const v16h*)wg_frag;
  v8f acc[4] = {};
#pragma unroll
  for (int ks = 0; ks < 8; ++ks) {
    v16h a = load_a_f32row(&agg[wid][r16][ks * 32], lane);
#pragma unroll
    for (int nt = 0; nt < 4; ++nt)
      acc[nt] = wmma_f16(a, bf[(ks * 4 + nt) * 32 + lane], acc[nt]);
  }
  size_t obase = (xbase + n0) * ENCc;
#pragma unroll
  for (int nt = 0; nt < 4; ++nt)
#pragma unroll
    for (int r = 0; r < 8; ++r) {
      int rowt = r + ((lane < 16) ? 0 : 8);
      xn[obase + (size_t)rowt * ENCc + nt * 16 + r16] = tanhf(acc[nt][r] * 0.25f);
    }
}

// ---------------------------------------------------------------- edge decoder: [640k,128]@[128,32] -> tanh -> @[32,1]
// 16-row tile == the 16 neighbors of one (b,n): b = tile/N, n = tile%N, d = lane&15
__global__ void decoder_kernel(const half_t* __restrict__ xh, const int* __restrict__ adj,
                               const int* __restrict__ numn, const half_t* __restrict__ wd1_frag,
                               const float* __restrict__ bd1, const float* __restrict__ wd2,
                               const float* __restrict__ bd2, float* __restrict__ nw) {
  int lane = threadIdx.x & 31, wid = threadIdx.x >> 5;
  int tile = blockIdx.x * 8 + wid;           // 5000*8 = 40000 tiles, exact
  int r16 = lane & 15;
  int b = tile / Nc;                          // uniform per wave
  int n = tile - b * Nc;                      // uniform per wave
  int d = r16;                                // A-row index == neighbor slot
  int av = adj[(b * Nc + n) * Dc + d];
  bool masked = (av == numn[b]);
  const half_t* selfh = xh + ((size_t)(b * (Nc + 1) + n)) * ENCc;   // shared by whole tile
  const half_t* nbrh  = xh + ((size_t)(b * (Nc + 1) + av)) * ENCc;
  __builtin_prefetch(nbrh, 0, 3);            // global_prefetch_b8 on the gathered row
  const v16h* bf = (const v16h*)wd1_frag;    // KS=4, NT=2
  v8f c0 = {}, c1 = {};
  const v16h zfrag = {};
#pragma unroll
  for (int ks = 0; ks < 4; ++ks) {
    v16h a;
    if (ks < 2) { a = load_a_f16row(selfh + ks * 32, lane); a = masked ? zfrag : a; }
    else        { a = load_a_f16row(nbrh + (ks - 2) * 32, lane); }
    c0 = wmma_f16(a, bf[(ks * 2 + 0) * 32 + lane], c0);
    c1 = wmma_f16(a, bf[(ks * 2 + 1) * 32 + lane], c1);
  }
  float w2a = wd2[r16], w2b = wd2[16 + r16];
  float b1a = bd1[r16], b1b = bd1[16 + r16];
#pragma unroll
  for (int r = 0; r < 8; ++r) {
    float t = tanhf(c0[r] + b1a) * w2a + tanhf(c1[r] + b1b) * w2b;
    t += __shfl_xor(t, 1); t += __shfl_xor(t, 2); t += __shfl_xor(t, 4); t += __shfl_xor(t, 8);
    if (r16 == 0) {
      int rowt = r + ((lane < 16) ? 0 : 8);
      nw[tile * 16 + rowt] = t + bd2[0];
    }
  }
}

__global__ void edge_softmax_kernel(float* __restrict__ nw, const int* __restrict__ adj,
                                    const int* __restrict__ numn) {
  int bn = blockIdx.x * blockDim.x + threadIdx.x;
  if (bn >= Bc * Nc) return;
  int nn = numn[bn / Nc];
  float v[Dc]; float mx = -3.0e38f;
  for (int d = 0; d < Dc; ++d) {
    float w = nw[bn * Dc + d];
    if (adj[bn * Dc + d] == nn) w -= BIGc;
    v[d] = w; mx = fmaxf(mx, w);
  }
  float s = 0.f;
  for (int d = 0; d < Dc; ++d) { float e2 = __expf(v[d] - mx); v[d] = e2; s += e2; }
  float inv = 1.f / s;
  for (int d = 0; d < Dc; ++d) nw[bn * Dc + d] = v[d] * inv;
}

// ---------------------------------------------------------------- dual decoder: [40k,64]@[64,32] -> tanh -> @[32,1]
__global__ void dual_dec_kernel(const half_t* __restrict__ xh, const half_t* __restrict__ wdd1_frag,
                                const float* __restrict__ bdd1, const float* __restrict__ wdd2,
                                const float* __restrict__ bdd2, float* __restrict__ dual_pad) {
  int lane = threadIdx.x & 31, wid = threadIdx.x >> 5;
  int tile = blockIdx.x * 8 + wid;
  if (tile >= Bc * TPB_NODE) return;        // wave-uniform
  int r16 = lane & 15;
  int b = tile / TPB_NODE;
  int n0 = (tile - b * TPB_NODE) * 16;
  const half_t* xr = xh + ((size_t)(b * (Nc + 1) + n0 + r16)) * ENCc;
  const v16h* bf = (const v16h*)wdd1_frag;   // KS=2, NT=2
  v8f c0 = {}, c1 = {};
#pragma unroll
  for (int ks = 0; ks < 2; ++ks) {
    v16h a = load_a_f16row(xr + ks * 32, lane);
    c0 = wmma_f16(a, bf[(ks * 2 + 0) * 32 + lane], c0);
    c1 = wmma_f16(a, bf[(ks * 2 + 1) * 32 + lane], c1);
  }
  float w2a = wdd2[r16], w2b = wdd2[16 + r16];
  float b1a = bdd1[r16], b1b = bdd1[16 + r16];
  size_t obase = (size_t)b * (Nc + 1) + n0;
#pragma unroll
  for (int r = 0; r < 8; ++r) {
    float t = tanhf(c0[r] + b1a) * w2a + tanhf(c1[r] + b1b) * w2b;
    t += __shfl_xor(t, 1); t += __shfl_xor(t, 2); t += __shfl_xor(t, 4); t += __shfl_xor(t, 8);
    if (r16 == 0) {
      int rowt = r + ((lane < 16) ? 0 : 8);
      dual_pad[obase + rowt] = t + bdd2[0];
    }
  }
}

// ---------------------------------------------------------------- flow solver
__global__ void flow_update_kernel(const float* __restrict__ nw, const float* __restrict__ inflow,
                                   const float* __restrict__ dem, float* __restrict__ flow) {
  int e = blockIdx.x * blockDim.x + threadIdx.x;  // 640000 exact
  int b = e / (Nc * Dc);
  int n = (e % (Nc * Dc)) / Dc;
  flow[e] = nw[e] * fmaxf(inflow[b * (Nc + 1) + n] - dem[b * Nc + n], 0.f);
}

__global__ void flow_scatter_kernel(const float* __restrict__ flow, const int* __restrict__ adj,
                                    float* __restrict__ inflow) {
  int e = blockIdx.x * blockDim.x + threadIdx.x;
  int b = e / (Nc * Dc);
  int rem = e % (Nc * Dc);
  int n = rem / Dc, d = rem % Dc;
  int av = adj[(b * Nc + n) * Dc + d];            // pad -> slot N (dropped on read)
  atomicAdd(&inflow[b * (Nc + 1) + av], flow[e]);
}

__global__ void flow_cost_kernel(const float* __restrict__ flow, float* __restrict__ out) {
  __shared__ float red[256];
  int tid = threadIdx.x;
  int e = blockIdx.x * 256 + tid;
  int b = e / (Nc * Dc);                           // uniform per block (256 | N*D)
  float v = flow[e];
  red[tid] = v * v;
  __syncthreads();
  for (int s = 128; s > 0; s >>= 1) { if (tid < s) red[tid] += red[tid + s]; __syncthreads(); }
  if (tid == 0) atomicAdd(&out[b], red[0]);
}

// dual flow: 20-step projected momentum descent, fully in-register; fold dual_cost into out
__global__ void dual_flow_kernel(const float* __restrict__ dual_pad, const int* __restrict__ adj,
                                 const int* __restrict__ numn, const float* __restrict__ dem,
                                 float* __restrict__ out) {
  __shared__ float red[256];
  int tid = threadIdx.x;
  int e = blockIdx.x * 256 + tid;
  int b = e / (Nc * Dc);
  int rem = e % (Nc * Dc);
  int n = rem / Dc, d = rem % Dc;
  int av = adj[(b * Nc + n) * Dc + d];
  float mk = (av == numn[b]) ? 0.f : 1.f;
  float dv  = dual_pad[b * (Nc + 1) + n];
  float dtr = dual_pad[b * (Nc + 1) + av];         // pad row = 0
  float diff = dtr - mk * dv;
  float f = 0.f, acc = 0.f;
#pragma unroll
  for (int it = 0; it < 20; ++it) {
    acc = 0.9f * acc + 0.1f * (2.f * f + diff);
    f = fmaxf(f - acc, 0.f) * mk;
  }
  float c = -(f * f + diff * f);                   // out -= dual_cost terms
  if (d == 0) c += dv * dem[b * Nc + n];           // out += dual_demand
  red[tid] = c;
  __syncthreads();
  for (int s = 128; s > 0; s >>= 1) { if (tid < s) red[tid] += red[tid + s]; __syncthreads(); }
  if (tid == 0) atomicAdd(&out[b], red[0]);
}

// ----------------------------------------------------------------
extern "C" void kernel_launch(void* const* d_in, const int* in_sizes, int n_in,
                              void* d_out, int out_size, void* d_ws, size_t ws_size,
                              hipStream_t stream) {
  (void)in_sizes; (void)n_in; (void)ws_size;
  const float* demands = (const float*)d_in[0];
  const float* feat    = (const float*)d_in[1];
  /* d_in[2] edge_lengths: unused by the reference math */
  const float* embv    = (const float*)d_in[3];
  const float* W_enc   = (const float*)d_in[4];
  const float* b_enc   = (const float*)d_in[5];
  const float* Wg      = (const float*)d_in[6];
  const float* a_src   = (const float*)d_in[7];
  const float* a_tgt   = (const float*)d_in[8];
  const float* Wd1     = (const float*)d_in[9];
  const float* bd1     = (const float*)d_in[10];
  const float* Wd2     = (const float*)d_in[11];
  const float* bd2     = (const float*)d_in[12];
  const float* Wdd1    = (const float*)d_in[13];
  const float* bdd1    = (const float*)d_in[14];
  const float* Wdd2    = (const float*)d_in[15];
  const float* bdd2    = (const float*)d_in[16];
  const int*   adj     = (const int*)d_in[17];
  const int*   numn    = (const int*)d_in[18];
  float* out = (float*)d_out;

  char* base = (char*)d_ws;
  size_t off = 0;
  auto alloc = [&](size_t bytes) { void* p = base + off; off = (off + bytes + 255) & ~(size_t)255; return p; };
  half_t* xin       = (half_t*)alloc((size_t)Bc * Nc * 64 * 2);
  half_t* xh        = (half_t*)alloc((size_t)Bc * (Nc + 1) * 64 * 2);
  half_t* wenc_frag = (half_t*)alloc(2 * 4 * 512 * 2);
  half_t* wg_frag   = (half_t*)alloc(8 * 4 * 512 * 2);
  half_t* wd1_frag  = (half_t*)alloc(4 * 2 * 512 * 2);
  half_t* wdd1_frag = (half_t*)alloc(2 * 2 * 512 * 2);
  float*  svec  = (float*)alloc(Hc * ENCc * 4);
  float*  tvec  = (float*)alloc(Hc * ENCc * 4);
  float*  x0    = (float*)alloc((size_t)Bc * (Nc + 1) * ENCc * 4);
  float*  x1    = (float*)alloc((size_t)Bc * (Nc + 1) * ENCc * 4);
  float*  s_src = (float*)alloc((size_t)Bc * Nc * Hc * 4);
  float*  s_tgt = (float*)alloc((size_t)Bc * (Nc + 1) * Hc * 4);
  float*  alpha = (float*)alloc((size_t)Bc * Nc * Dc * Hc * 4);
  float*  nw    = (float*)alloc((size_t)Bc * Nc * Dc * 4);
  float*  flow  = (float*)alloc((size_t)Bc * Nc * Dc * 4);
  float*  infl  = (float*)alloc((size_t)Bc * (Nc + 1) * 4);
  float*  dualp = (float*)alloc((size_t)Bc * (Nc + 1) * 4);

  prep_kernel<<<1, 256, 0, stream>>>(Wg, a_src, a_tgt, W_enc, Wd1, Wdd1,
                                     svec, tvec, wenc_frag, wg_frag, wd1_frag, wdd1_frag);
  build_xin_kernel<<<(Bc * Nc + 255) / 256, 256, 0, stream>>>(embv, feat, xin);
  zero_f_kernel<<<(Bc * (Nc + 1) + 255) / 256, 256, 0, stream>>>(infl, Bc * (Nc + 1));
  zero_f_kernel<<<1, 32, 0, stream>>>(out, out_size);
  zero_pads_kernel<<<3, 256, 0, stream>>>(x0, x1, s_tgt, dualp);

  encoder_kernel<<<313, 256, 0, stream>>>(xin, wenc_frag, b_enc, x0);

  float* xc = x0; float* xn = x1;
  for (int layer = 0; layer < 2; ++layer) {
    gat_scores_kernel<<<(Bc * Nc + 255) / 256, 256, 0, stream>>>(xc, svec, tvec, s_src, s_tgt);
    gat_attn_kernel<<<(Bc * Nc * Hc) / 256, 256, 0, stream>>>(s_src, s_tgt, adj, numn, alpha);
    gat_aggregate_kernel<<<1250, 64, 0, stream>>>(xc, alpha, adj, wg_frag, xn);
    float* tmp = xc; xc = xn; xn = tmp;
  }
  // after 2 layers xc == x0

  x_to_half_kernel<<<(Bc * (Nc + 1) * 8 + 255) / 256, 256, 0, stream>>>(xc, xh);
  decoder_kernel<<<5000, 256, 0, stream>>>(xh, adj, numn, wd1_frag, bd1, Wd2, bd2, nw);
  edge_softmax_kernel<<<(Bc * Nc + 255) / 256, 256, 0, stream>>>(nw, adj, numn);
  dual_dec_kernel<<<313, 256, 0, stream>>>(xh, wdd1_frag, bdd1, Wdd2, bdd2, dualp);

  for (int it = 0; it < 10; ++it) {
    flow_update_kernel<<<2500, 256, 0, stream>>>(nw, infl, demands, flow);
    if (it < 9) {
      zero_f_kernel<<<(Bc * (Nc + 1) + 255) / 256, 256, 0, stream>>>(infl, Bc * (Nc + 1));
      flow_scatter_kernel<<<2500, 256, 0, stream>>>(flow, adj, infl);
    }
  }
  flow_cost_kernel<<<2500, 256, 0, stream>>>(flow, out);
  dual_flow_kernel<<<2500, 256, 0, stream>>>(dualp, adj, numn, demands, out);
}
